// Vim_85736137162951
// MI455X (gfx1250) — matrix operand order repaired and checked by reading, not compiled, not run on hardware
//
#include <hip/hip_runtime.h>
#include <hip/hip_bf16.h>
#include <math.h>

// ---------- model constants ----------
#define B0     8
#define DEPTH  12
#define DIM    512
#define NSTATE 16
#define DTRANK 32
#define IMG    224
#define PATCH  16
#define CH     3
#define NCLS   1000
#define PD     768           // CH*PATCH*PATCH
#define GRID0  14            // IMG/PATCH
#define NPATCH 196           // GRID0*GRID0
#define SEQ    197           // NPATCH + cls
#define MROWS  (B0*SEQ)      // 1576
#define MPAT   (B0*NPATCH)   // 1568

typedef __attribute__((ext_vector_type(16))) __bf16 bf16x16;
typedef __attribute__((ext_vector_type(8)))  __bf16 bf16x8;
typedef __attribute__((ext_vector_type(8)))  float  f32x8;

// ---------- weight convert: W[L][K][N] (f32) -> WT[L][N][K] (bf16) ----------
__global__ void cvt_wT_kernel(const float* __restrict__ W, __bf16* __restrict__ WT,
                              int L, int K, int N) {
    int idx = blockIdx.x * blockDim.x + threadIdx.x;
    int total = L * K * N;
    if (idx >= total) return;
    int l = idx / (K * N);
    int rem = idx - l * (K * N);
    int n = rem / K;
    int k = rem - n * K;
    WT[idx] = (__bf16)W[(size_t)l * K * N + (size_t)k * N + n];
}

// ---------- im2col patches -> bf16 [B, 196, 768] ----------
__global__ void im2col_kernel(const float* __restrict__ x, __bf16* __restrict__ xp) {
    int idx = blockIdx.x * blockDim.x + threadIdx.x;
    if (idx >= B0 * NPATCH * PD) return;
    int b = idx / (NPATCH * PD);
    int rem = idx - b * (NPATCH * PD);
    int p = rem / PD;
    int q = rem - p * PD;
    int c  = q % CH;          // (p1*PATCH+p2)*CH + c
    int pp = q / CH;
    int p1 = pp / PATCH;
    int p2 = pp - p1 * PATCH;
    int gy = p / GRID0;
    int gx = p - gy * GRID0;
    size_t src = (((size_t)b * CH + c) * IMG + (gy * PATCH + p1)) * IMG + (gx * PATCH + p2);
    xp[idx] = (__bf16)x[src];
}

// ---------- WMMA GEMM: out[M,N] = act(A[M,K](bf16) @ WT[N,K]^T(bf16) + bias) ----------
// Register-blocked: one wave computes a 32x64 C tile (2 A frags x 4 B frags = 8 WMMA/K-step).
// act: 0=none, 1=silu, 2=softplus.  Requires N % 64 == 0, K % 32 == 0.
__global__ __launch_bounds__(256) void wmma_gemm_kernel(
    const __bf16* __restrict__ A, const __bf16* __restrict__ WT,
    const float* __restrict__ bias, float* __restrict__ out,
    int M, int N, int K, int act)
{
    int wave = threadIdx.x >> 5;
    int lane = threadIdx.x & 31;
    int tiles_n = N >> 6;              // 64-wide col tiles
    int tiles_m = (M + 31) >> 5;       // 32-tall row tiles
    int tile = blockIdx.x * 8 + wave;
    if (tile >= tiles_m * tiles_n) return;
    int tm = tile / tiles_n;
    int tn = tile - tm * tiles_n;
    int half = lane >> 4;
    int l15  = lane & 15;

    int r0 = tm * 32 + l15;       if (r0 >= M) r0 = M - 1;
    int r1 = tm * 32 + 16 + l15;  if (r1 >= M) r1 = M - 1;
    const __bf16* Ap0 = A + (size_t)r0 * K;
    const __bf16* Ap1 = A + (size_t)r1 * K;
    const __bf16* Bp[4];
#pragma unroll
    for (int j = 0; j < 4; ++j)
        Bp[j] = WT + (size_t)(tn * 64 + j * 16 + l15) * K;

    f32x8 c[2][4];
#pragma unroll
    for (int i = 0; i < 2; ++i)
#pragma unroll
        for (int j = 0; j < 4; ++j) c[i][j] = (f32x8){};

    for (int k0 = 0; k0 < K; k0 += 32) {
        // A fragments: per lane, K = k0 + half*8 + 0..7 and k0 + 16 + half*8 + 0..7
        union { bf16x16 v; bf16x8 h[2]; } a0, a1;
        a0.h[0] = *(const bf16x8*)(Ap0 + k0 + half * 8);
        a0.h[1] = *(const bf16x8*)(Ap0 + k0 + 16 + half * 8);
        a1.h[0] = *(const bf16x8*)(Ap1 + k0 + half * 8);
        a1.h[1] = *(const bf16x8*)(Ap1 + k0 + 16 + half * 8);
        // B fragments: column n holds K = k0 + half*8 + 0..7 and k0 + 16 + half*8 + 0..7
        bf16x16 bf[4];
#pragma unroll
        for (int j = 0; j < 4; ++j) {
            union { bf16x16 v; bf16x8 h[2]; } bt;
            bt.h[0] = *(const bf16x8*)(Bp[j] + k0 + half * 8);
            bt.h[1] = *(const bf16x8*)(Bp[j] + k0 + 16 + half * 8);
            bf[j] = bt.v;
        }
        if (k0 + 256 < K) {
            __builtin_prefetch(Bp[0] + k0 + 256, 0, 1);
            __builtin_prefetch(Bp[2] + k0 + 256, 0, 1);
        }
#pragma unroll
        for (int j = 0; j < 4; ++j) {
            c[0][j] = __builtin_amdgcn_wmma_f32_16x16x32_bf16(
                          false, a0.v, false, bf[j], (short)0, c[0][j], false, false);
            c[1][j] = __builtin_amdgcn_wmma_f32_16x16x32_bf16(
                          false, a1.v, false, bf[j], (short)0, c[1][j], false, false);
        }
    }

#pragma unroll
    for (int j = 0; j < 4; ++j) {
        int n = tn * 64 + j * 16 + l15;
        float bv = bias ? bias[n] : 0.0f;
#pragma unroll
        for (int i = 0; i < 2; ++i) {
#pragma unroll
            for (int r = 0; r < 8; ++r) {
                int m = tm * 32 + i * 16 + half * 8 + r;
                if (m < M) {
                    float v = c[i][j][r] + bv;
                    if (act == 1)      v = v / (1.0f + __expf(-v));               // silu
                    else if (act == 2) v = (v > 20.0f) ? v : log1pf(__expf(v));   // softplus
                    out[(size_t)m * N + n] = v;
                }
            }
        }
    }
}

// ---------- assemble h: cls token row + patch-embed rows ----------
__global__ void assemble_h_kernel(const float* __restrict__ hemb,
                                  const float* __restrict__ cls,
                                  float* __restrict__ h) {
    int idx = blockIdx.x * blockDim.x + threadIdx.x;
    if (idx >= MROWS * DIM) return;
    int row = idx / DIM;
    int d = idx - row * DIM;
    int b = row / SEQ;
    int l = row - b * SEQ;
    h[idx] = (l == 0) ? cls[d] : hemb[((size_t)b * NPATCH + (l - 1)) * DIM + d];
}

// ---------- layernorm over last dim (cols), optional bf16 and/or f32 output ----------
__global__ __launch_bounds__(128) void ln_kernel(
    const float* __restrict__ x, const float* __restrict__ g, const float* __restrict__ b,
    __bf16* __restrict__ out_bf, float* __restrict__ out_f, int cols)
{
    __shared__ float s1[128], s2[128];
    int row = blockIdx.x;
    const float* xr = x + (size_t)row * cols;
    float sum = 0.f, sq = 0.f;
    for (int i = threadIdx.x; i < cols; i += 128) { float v = xr[i]; sum += v; sq += v * v; }
    s1[threadIdx.x] = sum; s2[threadIdx.x] = sq;
    __syncthreads();
    for (int st = 64; st > 0; st >>= 1) {
        if (threadIdx.x < st) { s1[threadIdx.x] += s1[threadIdx.x + st];
                                s2[threadIdx.x] += s2[threadIdx.x + st]; }
        __syncthreads();
    }
    float mean = s1[0] / cols;
    float var  = s2[0] / cols - mean * mean;
    float rstd = rsqrtf(var + 1e-5f);
    for (int i = threadIdx.x; i < cols; i += 128) {
        float v = (xr[i] - mean) * rstd * g[i] + b[i];
        if (out_bf) out_bf[(size_t)row * cols + i] = (__bf16)v;
        if (out_f)  out_f [(size_t)row * cols + i] = v;
    }
}

// ---------- dbc = u @ dbc_w  (K=512, N=64) ----------
__global__ void dbc_kernel(const float* __restrict__ u, const float* __restrict__ w,
                           float* __restrict__ dbc) {
    int idx = blockIdx.x * blockDim.x + threadIdx.x;
    if (idx >= MROWS * (DTRANK + 2 * NSTATE)) return;
    const int NO = DTRANK + 2 * NSTATE;  // 64
    int m = idx / NO;
    int j = idx - m * NO;
    const float* ur = u + (size_t)m * DIM;
    float acc = 0.f;
    for (int k = 0; k < DIM; ++k) acc = fmaf(ur[k], w[(size_t)k * NO + j], acc);
    dbc[idx] = acc;
}

// ---------- delta = softplus(dt @ dtp_w + dtp_b)  (K=32, N=512) ----------
__global__ void delta_kernel(const float* __restrict__ dbc, const float* __restrict__ w,
                             const float* __restrict__ bias, float* __restrict__ delta) {
    int idx = blockIdx.x * blockDim.x + threadIdx.x;
    if (idx >= MROWS * DIM) return;
    const int NO = DTRANK + 2 * NSTATE;
    int m = idx / DIM;
    int d = idx - m * DIM;
    const float* dt = dbc + (size_t)m * NO;
    float acc = bias[d];
#pragma unroll
    for (int r = 0; r < DTRANK; ++r) acc = fmaf(dt[r], w[(size_t)r * DIM + d], acc);
    delta[idx] = (acc > 20.0f) ? acc : log1pf(__expf(acc));
}

// ---------- selective scan: one thread per (batch, channel), 16-state in regs ----------
__global__ __launch_bounds__(256) void scan_kernel(
    const float* __restrict__ u, const float* __restrict__ delta,
    const float* __restrict__ dbc, const float* __restrict__ A_log,
    const float* __restrict__ Dp, float* __restrict__ yacc, int accumulate)
{
    int idx = blockIdx.x * blockDim.x + threadIdx.x;
    if (idx >= B0 * DIM) return;
    int b = idx / DIM;
    int d = idx - b * DIM;
    const int NO = DTRANK + 2 * NSTATE;
    float A[NSTATE], s[NSTATE];
#pragma unroll
    for (int n = 0; n < NSTATE; ++n) { A[n] = -__expf(A_log[(size_t)d * NSTATE + n]); s[n] = 0.f; }
    float Dd = Dp[d];
    for (int t = 0; t < SEQ; ++t) {
        size_t m = (size_t)b * SEQ + t;
        float dl = delta[m * DIM + d];
        float uv = u[m * DIM + d];
        const float* BC = dbc + m * NO;
        float y = 0.f;
#pragma unroll
        for (int n = 0; n < NSTATE; ++n) {
            float dA = __expf(dl * A[n]);
            s[n] = fmaf(dA, s[n], dl * BC[DTRANK + n] * uv);
            y = fmaf(s[n], BC[DTRANK + NSTATE + n], y);
        }
        y = fmaf(uv, Dd, y);
        float* yp = yacc + m * DIM + d;
        *yp = accumulate ? (*yp + y) : y;
    }
}

// ---------- residual combine: h = yacc * z + h ----------
__global__ void residual_kernel(const float* __restrict__ yacc, const float* __restrict__ z,
                                float* __restrict__ h) {
    int idx = blockIdx.x * blockDim.x + threadIdx.x;
    if (idx >= MROWS * DIM) return;
    h[idx] = fmaf(yacc[idx], z[idx], h[idx]);
}

// ---------- pool: mean over patch rows (drop cls) ----------
__global__ void pool_kernel(const float* __restrict__ h, float* __restrict__ pooled) {
    int idx = blockIdx.x * blockDim.x + threadIdx.x;
    if (idx >= B0 * DIM) return;
    int b = idx / DIM;
    int d = idx - b * DIM;
    float acc = 0.f;
    for (int l = 1; l < SEQ; ++l) acc += h[((size_t)b * SEQ + l) * DIM + d];
    pooled[idx] = acc * (1.0f / NPATCH);
}

// ---------- head: out = pooled_n @ head_w + head_b ----------
__global__ void head_kernel(const float* __restrict__ p, const float* __restrict__ w,
                            const float* __restrict__ bias, float* __restrict__ out) {
    int idx = blockIdx.x * blockDim.x + threadIdx.x;
    if (idx >= B0 * NCLS) return;
    int b = idx / NCLS;
    int c = idx - b * NCLS;
    const float* pr = p + (size_t)b * DIM;
    float acc = bias[c];
    for (int d = 0; d < DIM; ++d) acc = fmaf(pr[d], w[(size_t)d * NCLS + c], acc);
    out[idx] = acc;
}

// =====================================================================
extern "C" void kernel_launch(void* const* d_in, const int* in_sizes, int n_in,
                              void* d_out, int out_size, void* d_ws, size_t ws_size,
                              hipStream_t stream) {
    const float* x        = (const float*)d_in[0];
    const float* patch_w  = (const float*)d_in[1];
    const float* patch_b  = (const float*)d_in[2];
    const float* cls_tok  = (const float*)d_in[3];
    const float* fc_w     = (const float*)d_in[4];
    const float* fc_b     = (const float*)d_in[5];
    const float* bc_w     = (const float*)d_in[6];
    const float* bc_b     = (const float*)d_in[7];
    const float* ln_g     = (const float*)d_in[8];
    const float* ln_b     = (const float*)d_in[9];
    const float* proj_w   = (const float*)d_in[10];
    const float* proj_b   = (const float*)d_in[11];
    const float* dbc_w    = (const float*)d_in[12];
    const float* dtp_w    = (const float*)d_in[13];
    const float* dtp_b    = (const float*)d_in[14];
    const float* A_log    = (const float*)d_in[15];
    const float* D_param  = (const float*)d_in[16];
    const float* hln_g    = (const float*)d_in[17];
    const float* hln_b    = (const float*)d_in[18];
    const float* head_w   = (const float*)d_in[19];
    const float* head_b   = (const float*)d_in[20];
    float* out = (float*)d_out;

    // workspace carve-up
    char* base = (char*)d_ws;
    size_t off = 0;
    auto alloc = [&](size_t bytes) -> char* {
        char* p = base + off;
        off = (off + bytes + 255) & ~(size_t)255;
        return p;
    };
    __bf16* wt_fc    = (__bf16*)alloc((size_t)DEPTH * DIM * DIM * 2);
    __bf16* wt_bc    = (__bf16*)alloc((size_t)DEPTH * DIM * DIM * 2);
    __bf16* wt_proj  = (__bf16*)alloc((size_t)DEPTH * DIM * DIM * 2);
    __bf16* wt_patch = (__bf16*)alloc((size_t)DIM * PD * 2);
    __bf16* xp       = (__bf16*)alloc((size_t)MPAT * PD * 2);
    float*  hemb     = (float*) alloc((size_t)MPAT * DIM * 4);
    float*  h        = (float*) alloc((size_t)MROWS * DIM * 4);
    __bf16* xn       = (__bf16*)alloc((size_t)MROWS * DIM * 2);
    float*  z        = (float*) alloc((size_t)MROWS * DIM * 4);
    float*  u        = (float*) alloc((size_t)MROWS * DIM * 4);
    float*  dbcb     = (float*) alloc((size_t)MROWS * 64 * 4);
    float*  delta    = (float*) alloc((size_t)MROWS * DIM * 4);
    float*  yacc     = (float*) alloc((size_t)MROWS * DIM * 4);
    float*  pooled   = (float*) alloc((size_t)B0 * DIM * 4);
    float*  pooledn  = (float*) alloc((size_t)B0 * DIM * 4);
    (void)ws_size; (void)in_sizes; (void)n_in; (void)out_size;

    const int T = 256;
    auto cdiv = [](int a, int b) { return (a + b - 1) / b; };

    // 1. convert weights to transposed bf16
    cvt_wT_kernel<<<cdiv(DEPTH*DIM*DIM, T), T, 0, stream>>>(fc_w,   wt_fc,   DEPTH, DIM, DIM);
    cvt_wT_kernel<<<cdiv(DEPTH*DIM*DIM, T), T, 0, stream>>>(bc_w,   wt_bc,   DEPTH, DIM, DIM);
    cvt_wT_kernel<<<cdiv(DEPTH*DIM*DIM, T), T, 0, stream>>>(proj_w, wt_proj, DEPTH, DIM, DIM);
    cvt_wT_kernel<<<cdiv(PD*DIM, T), T, 0, stream>>>(patch_w, wt_patch, 1, PD, DIM);

    // 2. patch embed
    im2col_kernel<<<cdiv(B0*NPATCH*PD, T), T, 0, stream>>>(x, xp);
    {
        int tiles = cdiv(MPAT, 32) * (DIM / 64);   // 49*8 = 392
        wmma_gemm_kernel<<<cdiv(tiles, 8), T, 0, stream>>>(xp, wt_patch, patch_b, hemb,
                                                           MPAT, DIM, PD, 0);
    }
    assemble_h_kernel<<<cdiv(MROWS*DIM, T), T, 0, stream>>>(hemb, cls_tok, h);

    // 3. layers
    int tilesL = cdiv(MROWS, 32) * (DIM / 64);     // 50*8 = 400
    for (int l = 0; l < DEPTH; ++l) {
        const size_t wOff  = (size_t)l * DIM * DIM;
        ln_kernel<<<MROWS, 128, 0, stream>>>(h, ln_g + l*DIM, ln_b + l*DIM, xn, nullptr, DIM);
        // z = silu(xn @ proj + b)
        wmma_gemm_kernel<<<cdiv(tilesL, 8), T, 0, stream>>>(xn, wt_proj + wOff,
                                                            proj_b + l*DIM, z,
                                                            MROWS, DIM, DIM, 1);
        for (int branch = 0; branch < 2; ++branch) {
            const __bf16* wt = branch ? wt_bc : wt_fc;
            const float*  wb = branch ? (bc_b + l*DIM) : (fc_b + l*DIM);
            // u = softplus(xn @ W + b)
            wmma_gemm_kernel<<<cdiv(tilesL, 8), T, 0, stream>>>(xn, wt + wOff, wb, u,
                                                                MROWS, DIM, DIM, 2);
            dbc_kernel<<<cdiv(MROWS*64, T), T, 0, stream>>>(u, dbc_w + (size_t)l*DIM*64, dbcb);
            delta_kernel<<<cdiv(MROWS*DIM, T), T, 0, stream>>>(dbcb,
                                                               dtp_w + (size_t)l*DTRANK*DIM,
                                                               dtp_b + l*DIM, delta);
            scan_kernel<<<cdiv(B0*DIM, T), T, 0, stream>>>(u, delta, dbcb,
                                                           A_log + (size_t)l*DIM*NSTATE,
                                                           D_param + l*DIM, yacc, branch);
        }
        residual_kernel<<<cdiv(MROWS*DIM, T), T, 0, stream>>>(yacc, z, h);
    }

    // 4. head
    pool_kernel<<<cdiv(B0*DIM, T), T, 0, stream>>>(h, pooled);
    ln_kernel<<<B0, 128, 0, stream>>>(pooled, hln_g, hln_b, nullptr, pooledn, DIM);
    head_kernel<<<cdiv(B0*NCLS, T), T, 0, stream>>>(pooledn, head_w, head_b, out);
}
